// ASR_cnn_60052232733187
// MI455X (gfx1250) — compile-verified, matrix operand order checked by reading, not run on hardware
//
#include <hip/hip_runtime.h>
#include <hip/hip_bf16.h>

// ---------------------------------------------------------------------------
// Problem constants (from the reference)
// ---------------------------------------------------------------------------
#define CIN   80
#define CH    256
#define TLEN  2000
#define BATCH 32
#define NLAB  32
#define EPSV  1e-5f

typedef __attribute__((ext_vector_type(16))) __bf16 v16bf;
typedef __attribute__((ext_vector_type(8)))  __bf16 v8bf;
typedef __attribute__((ext_vector_type(8)))  float  v8f;

union BF16x16 { v16bf v; v8bf h[2]; };

// ---------------------------------------------------------------------------
// Kernel 1: fold BN into conv weights, convert to bf16, pack in WMMA-A layout.
//   apack[(k*8 + chunk)*256 + co][ci&31]  (32 contiguous bf16 per (k,chunk,co))
// One launch per rb layer. idx = k*65536 + co*256 + ci.
// ---------------------------------------------------------------------------
__global__ void prep_rb_kernel(const float* __restrict__ w,
                               const float* __restrict__ bb,
                               const float* __restrict__ gamma,
                               const float* __restrict__ beta,
                               const float* __restrict__ mean,
                               const float* __restrict__ var,
                               __bf16* __restrict__ apack,
                               float*  __restrict__ bias) {
  int idx = blockIdx.x * 256 + threadIdx.x;           // 0 .. 7*256*256-1
  int ci  = idx & 255;
  int co  = (idx >> 8) & 255;
  int k   = idx >> 16;                                 // 0..6
  float scale = gamma[co] * rsqrtf(var[co] + EPSV);
  float wf = w[(co * CH + ci) * 7 + k] * scale;
  int chunk = ci >> 5, cil = ci & 31;
  apack[(((k * 8 + chunk) * CH) + co) * 32 + cil] = (__bf16)wf;
  if (k == 0 && ci == 0)
    bias[co] = (bb[co] - mean[co]) * scale + beta[co];
}

// ---------------------------------------------------------------------------
// Kernel 2: entry 1x1 conv (80->256) + BN + tanh, FP32 scalar (BW-trivial).
// ---------------------------------------------------------------------------
__global__ void extra_conv_kernel(const float* __restrict__ x,
                                  const float* __restrict__ w,
                                  const float* __restrict__ bb,
                                  const float* __restrict__ gamma,
                                  const float* __restrict__ beta,
                                  const float* __restrict__ mean,
                                  const float* __restrict__ var,
                                  float* __restrict__ H) {
  int idx = blockIdx.x * 256 + threadIdx.x;           // b*CH*T + co*T + t
  int t  = idx % TLEN;
  int co = (idx / TLEN) % CH;
  int b  = idx / (TLEN * CH);
  const float* xp = x + (size_t)b * CIN * TLEN + t;
  const float* wp = w + co * CIN;
  float acc = 0.f;
#pragma unroll 8
  for (int ci = 0; ci < CIN; ++ci) acc += xp[ci * TLEN] * wp[ci];
  float scale = gamma[co] * rsqrtf(var[co] + EPSV);
  float y = (acc + bb[co] - mean[co]) * scale + beta[co];
  H[idx] = tanhf(y);
}

// ---------------------------------------------------------------------------
// Kernel 3: dilated conv (k=7, 256->256) via bf16 WMMA + gated-tanh + residual.
// Block = 128 threads = 4 waves; each wave computes a 16(co) x (NT*16)(t) strip
// with NT f32 accumulators, so each A (weight) fragment is loaded once and
// reused NT times.  NT is chosen per dilation so the staged LDS tile
// (16*NT + 6D) rows x 528 B stays under 64 KB.
// LDS tile: rows = t_local, cols = 256 ci (+8 pad -> 528 B rows, conflict-free
// 16-lane b128 reads).  A stream: one lane-constant base + immediate offsets.
// SKIP_MODE: 0 = none, 1 = skip = ha, 2 = skip += ha.
// ---------------------------------------------------------------------------
template <int D, int SKIP_MODE>
__global__ __launch_bounds__(128) void rb_conv_kernel(
    const float* __restrict__ Hin, float* __restrict__ Hout,
    const __bf16* __restrict__ apack, const float* __restrict__ bias,
    float* __restrict__ skip) {
  constexpr int NTRAW = (124 - 6 * D) / 16;
  constexpr int NT  = NTRAW < 1 ? 1 : NTRAW;   // 16-col sub-tiles per wave
  constexpr int TT  = 16 * NT;                 // t extent per block
  constexpr int TW  = TT + 6 * D;              // staged t extent (incl. halo)
  constexpr int ROW = CH + 8;                  // padded LDS row (528 B)
  __shared__ __align__(16) __bf16 lds[TW * ROW];

  const int tid = threadIdx.x;
  const int t0  = blockIdx.x * TT;
  const int b   = blockIdx.z;

  // ---- cooperative stage: H[b, :, t0-3D .. t0+TT-1+3D] -> LDS (bf16, t-major)
  const float* hb = Hin + (size_t)b * CH * TLEN;
  for (int i = tid; i < TW * CH; i += 128) {
    int ci = i / TW;
    int tl = i - ci * TW;
    int gt = t0 - 3 * D + tl;
    float v = (gt >= 0 && gt < TLEN) ? hb[ci * TLEN + gt] : 0.0f;
    lds[tl * ROW + ci] = (__bf16)v;
  }
  __syncthreads();

  const int wave = tid >> 5;
  const int lane = tid & 31;
  const int g    = lane >> 4;   // K-half selector
  const int m    = lane & 15;   // A: M row / B: N column
  const int co0  = (blockIdx.y * 4 + wave) * 16;

  // C initialized with folded bias: element r maps to co = co0 + r + 8*g.
  v8f acc[NT];
#pragma unroll
  for (int n = 0; n < NT; ++n)
#pragma unroll
    for (int r = 0; r < 8; ++r) acc[n][r] = bias[co0 + r + 8 * g];

  // Lane-constant bases; all inner offsets are compile-time immediates.
  const __bf16* abase = apack + (co0 + m) * 32 + g * 8;
  const __bf16* bbase = lds + m * ROW + g * 8;

#pragma unroll
  for (int k = 0; k < 7; ++k) {
#pragma unroll
    for (int chunk = 0; chunk < 8; ++chunk) {
      const __bf16* ar = abase + (k * 8 + chunk) * (CH * 32);
      BF16x16 ua;
      ua.h[0] = *(const v8bf*)(ar);
      ua.h[1] = *(const v8bf*)(ar + 16);
#pragma unroll
      for (int n = 0; n < NT; ++n) {
        const __bf16* br = bbase + (k * D + n * 16) * ROW + chunk * 32;
        BF16x16 ub;
        ub.h[0] = *(const v8bf*)(br);
        ub.h[1] = *(const v8bf*)(br + 16);
        acc[n] = __builtin_amdgcn_wmma_f32_16x16x32_bf16(
            false, ua.v, false, ub.v, (short)0, acc[n], false, false);
      }
    }
  }

  // ---- epilogue: ha = tanh(tanh(y)*sigmoid(y)); h' = ha + h; skip ops
#pragma unroll
  for (int n = 0; n < NT; ++n) {
    int t = t0 + n * 16 + m;
    if (t < TLEN) {
#pragma unroll
      for (int r = 0; r < 8; ++r) {
        int co = co0 + r + 8 * g;
        float y  = acc[n][r];
        float th = tanhf(y);
        float sg = 1.0f / (1.0f + expf(-y));
        float ha = tanhf(th * sg);
        size_t idx = ((size_t)b * CH + co) * TLEN + t;
        Hout[idx] = ha + Hin[idx];
        if (SKIP_MODE == 1)      skip[idx] = ha;
        else if (SKIP_MODE == 2) skip[idx] += ha;
      }
    }
  }
}

// ---------------------------------------------------------------------------
// Kernel 4: relu(shortcut) -> 1x1 conv (256->32) + BN + log_softmax,
// transposed output [T, B, 32]. One wave per (b,t); lane = label.
// ---------------------------------------------------------------------------
__global__ void out_conv_kernel(const float* __restrict__ skip,
                                const float* __restrict__ w,
                                const float* __restrict__ bb,
                                const float* __restrict__ gamma,
                                const float* __restrict__ beta,
                                const float* __restrict__ mean,
                                const float* __restrict__ var,
                                float* __restrict__ out) {
  int gw   = blockIdx.x * 8 + (threadIdx.x >> 5);   // global wave id, b*T + t
  int lane = threadIdx.x & 31;                      // label index
  int b = gw / TLEN, t = gw - b * TLEN;
  const float* sp = skip + (size_t)b * CH * TLEN + t;
  const float* wp = w + lane * CH;
  float acc = 0.f;
#pragma unroll 4
  for (int ci = 0; ci < CH; ++ci) {
    float s = sp[ci * TLEN];
    s = s > 0.f ? s : 0.f;
    acc += s * wp[ci];
  }
  float scale = gamma[lane] * rsqrtf(var[lane] + EPSV);
  float y = (acc + bb[lane] - mean[lane]) * scale + beta[lane];
  float mx = y;
#pragma unroll
  for (int off = 16; off; off >>= 1) mx = fmaxf(mx, __shfl_xor(mx, off, 32));
  float e = expf(y - mx);
  float s = e;
#pragma unroll
  for (int off = 16; off; off >>= 1) s += __shfl_xor(s, off, 32);
  out[((size_t)t * BATCH + b) * NLAB + lane] = (y - mx) - logf(s);
}

// ---------------------------------------------------------------------------
// Host side
// ---------------------------------------------------------------------------
static inline int nt_for(int d) {
  int n = (124 - 6 * d) / 16;
  return n < 1 ? 1 : n;
}

static void launch_rb(int l, int mode, const float* Hi, float* Ho,
                      const __bf16* ap, const float* bias, float* skip,
                      hipStream_t s) {
  static const int DIL[5] = {1, 2, 4, 8, 16};
  int tt = 16 * nt_for(DIL[l]);
  dim3 grid((TLEN + tt - 1) / tt, 4, BATCH);
  dim3 blk(128);
#define RB_CASE(DV)                                                            \
  do {                                                                         \
    if (mode == 0)                                                             \
      rb_conv_kernel<DV, 0><<<grid, blk, 0, s>>>(Hi, Ho, ap, bias, skip);      \
    else if (mode == 1)                                                        \
      rb_conv_kernel<DV, 1><<<grid, blk, 0, s>>>(Hi, Ho, ap, bias, skip);      \
    else                                                                       \
      rb_conv_kernel<DV, 2><<<grid, blk, 0, s>>>(Hi, Ho, ap, bias, skip);      \
  } while (0)
  switch (l) {
    case 0: RB_CASE(1);  break;
    case 1: RB_CASE(2);  break;
    case 2: RB_CASE(4);  break;
    case 3: RB_CASE(8);  break;
    default: RB_CASE(16); break;
  }
#undef RB_CASE
}

extern "C" void kernel_launch(void* const* d_in, const int* in_sizes, int n_in,
                              void* d_out, int out_size, void* d_ws, size_t ws_size,
                              hipStream_t stream) {
  (void)n_in; (void)out_size; (void)ws_size;

  // ---- input pointer mapping; internal per-set order: w,b,gamma,beta,mean,var
  const float* x = nullptr;
  const float* extraP[6];
  const float* rbP[5][6];
  const float* outP[6];
  bool alphabetical = (in_sizes[0] == 1);   // jax tree_flatten puts num_blocks first
  if (!alphabetical) {
    // insertion order: x, extra{w,b,g,be,m,v}, rb[0..4]{...}, out{...}, num_blocks
    x = (const float*)d_in[0];
    for (int j = 0; j < 6; ++j) extraP[j] = (const float*)d_in[1 + j];
    for (int l = 0; l < 5; ++l)
      for (int j = 0; j < 6; ++j) rbP[l][j] = (const float*)d_in[7 + 6 * l + j];
    for (int j = 0; j < 6; ++j) outP[j] = (const float*)d_in[37 + j];
  } else {
    // sorted: num_blocks, params{extra,out,rb[0..4]} each {b,beta,gamma,mean,var,w}, x
    static const int perm[6] = {5, 0, 2, 1, 3, 4};  // internal j -> sorted offset
    for (int j = 0; j < 6; ++j) extraP[j] = (const float*)d_in[1 + perm[j]];
    for (int j = 0; j < 6; ++j) outP[j]   = (const float*)d_in[7 + perm[j]];
    for (int l = 0; l < 5; ++l)
      for (int j = 0; j < 6; ++j)
        rbP[l][j] = (const float*)d_in[13 + 6 * l + perm[j]];
    x = (const float*)d_in[43];
  }

  // ---- workspace layout (all offsets 256B-aligned)
  const size_t HSZ   = (size_t)BATCH * CH * TLEN * sizeof(float);   // 65,536,000
  const size_t APELT = (size_t)5 * 7 * 8 * CH * 32;                 // bf16 elems
  char* ws = (char*)d_ws;
  float*  H0    = (float*)(ws);
  float*  H1    = (float*)(ws + HSZ);
  float*  skip  = (float*)(ws + 2 * HSZ);
  __bf16* apack = (__bf16*)(ws + 3 * HSZ);
  float*  bias  = (float*)(ws + 3 * HSZ + ((APELT * 2 + 255) & ~(size_t)255));

  // ---- 1) fold BN + pack bf16 weights (5 layers)
  for (int l = 0; l < 5; ++l) {
    prep_rb_kernel<<<(7 * CH * CH) / 256, 256, 0, stream>>>(
        rbP[l][0], rbP[l][1], rbP[l][2], rbP[l][3], rbP[l][4], rbP[l][5],
        apack + (size_t)l * 7 * 8 * CH * 32, bias + l * CH);
  }

  // ---- 2) entry conv + tanh -> H0
  extra_conv_kernel<<<(BATCH * CH * TLEN) / 256, 256, 0, stream>>>(
      x, extraP[0], extraP[1], extraP[2], extraP[3], extraP[4], extraP[5], H0);

  // ---- 3) 3 blocks x 5 dilated WMMA convs, ping-pong H buffers
  const float* Hi = H0;
  float*       Ho = H1;
  for (int blk = 0; blk < 3; ++blk) {        // num_blocks = 3 (per setup_inputs)
    for (int l = 0; l < 5; ++l) {
      int mode = (blk == 2) ? (l == 0 ? 1 : 2) : 0;  // only last block's skips live
      launch_rb(l, mode, Hi, Ho,
                apack + (size_t)l * 7 * 8 * CH * 32, bias + l * CH, skip, stream);
      const float* tmp = Ho; Ho = (float*)Hi; Hi = tmp;
    }
  }

  // ---- 4) relu + 1x1 conv + log_softmax -> [T, B, 32]
  out_conv_kernel<<<(BATCH * TLEN) / 8, 256, 0, stream>>>(
      skip, outP[0], outP[1], outP[2], outP[3], outP[4], outP[5],
      (float*)d_out);
}
